// LittleBitLinearHF_20856361190081
// MI455X (gfx1250) — matrix operand order repaired and checked by reading, not compiled, not run on hardware
//
#include <hip/hip_runtime.h>
#include <hip/hip_bf16.h>

// y = ((x*g) @ sign(V)) * ell @ sign(U)^T * h + bias
// ROWS=8192, D_IN=4096, D_OUT=4096, R=128. HBM-bound (~272 MB -> ~11.7us @23.3TB/s).

typedef __attribute__((ext_vector_type(16))) _Float16 v16h;
typedef __attribute__((ext_vector_type(8)))  _Float16 v8h;
typedef __attribute__((ext_vector_type(2)))  _Float16 v2h;
typedef __attribute__((ext_vector_type(8)))  float    v8f;
typedef __attribute__((ext_vector_type(2)))  float    v2f;
typedef __attribute__((ext_vector_type(4)))  unsigned int v4u;
typedef __attribute__((ext_vector_type(8)))  int      v8i_t;
typedef __attribute__((ext_vector_type(4)))  int      v4i_t;

#define ROWS   8192
#define D_IN   4096
#define D_OUT  4096
#define RANK   128

// ---------------------------------------------------------------------------
// Kernel 0: fp32 factors -> exact +/-1 f16, PRE-SWIZZLED into WMMA B-fragment
// order: fragment (k-chunk, 16-col slice) = 32 lanes x 16 contiguous halfs.
//   B layout: lane = (k%32)/16*16 + col%16 ; element j -> K = k0 + 16*(lane/16) + j
//   Vs_sw [(kc*8   + cslice)*512 + lane*16 + j],  kc = k/32 over D_IN
//   UT_sw [(kc*256 + cslice)*512 + lane*16 + j],  kc = k/32 over RANK
// ---------------------------------------------------------------------------
__global__ void lbl_sign_convert(const float* __restrict__ U_fp,
                                 const float* __restrict__ V_fp,
                                 _Float16* __restrict__ Vs_sw,
                                 _Float16* __restrict__ UT_sw) {
    int i = blockIdx.x * blockDim.x + threadIdx.x;   // [0, 4096*128)

    // ---- V path: V_fp[k][n], k over D_IN, n over RANK ----
    {
        int k = i >> 7, n = i & (RANK - 1);
        int kc = k >> 5, k2 = k & 31;
        int half = k2 >> 4, j = k2 & 15;
        int lane = half * 16 + (n & 15);
        int cs   = n >> 4;
        Vs_sw[(size_t)((kc * 8 + cs) << 9) + lane * 16 + j] =
            (_Float16)(V_fp[i] >= 0.0f ? 1.0f : -1.0f);
    }
    // ---- U path: U_fp[n][k], n over D_OUT, k over RANK; B = U^T ----
    {
        int n = i >> 7, k = i & (RANK - 1);
        int kc = k >> 5, k2 = k & 31;
        int half = k2 >> 4, j = k2 & 15;
        int lane = half * 16 + (n & 15);
        int cs   = n >> 4;                            // 0..255
        UT_sw[(size_t)((kc * 256 + cs) << 9) + lane * 16 + j] =
            (_Float16)(U_fp[i] >= 0.0f ? 1.0f : -1.0f);
    }
}

// ---------------------------------------------------------------------------
// Helper: build v16h A fragment from a row-major padded LDS panel.
// a[j] = lds[row*stride + kbase + half*16 + j], a[8+j] = same +8 halfs.
// Both runs are 8 contiguous, 16B-aligned halfs -> two ds_load_b128.
// ---------------------------------------------------------------------------
__device__ __forceinline__ v16h load_a_frag(const _Float16* lds, int row,
                                            int stride, int kbase, int half) {
    const v8h alo = *(const v8h*)&lds[row * stride + kbase + half * 16];
    const v8h ahi = *(const v8h*)&lds[row * stride + kbase + 8 + half * 16];
    return __builtin_shufflevector(alo, ahi, 0, 1, 2, 3, 4, 5, 6, 7,
                                   8, 9, 10, 11, 12, 13, 14, 15);
}

// ---------------------------------------------------------------------------
// Kernel 1: T[ROWS x RANK] (f16, ell applied) = (x*g) @ sign(V)
// Block = 8 waves = 16-row tile; LDS shares the 16x32 (x*g) f16 panel.
// ---------------------------------------------------------------------------
#define S1_STRIDE 40   // halfs; 80B row stride (16B aligned, conflict-spread)

__global__ void lbl_stage1(const float* __restrict__ x,
                           const float* __restrict__ g,
                           const _Float16* __restrict__ Vs_sw,
                           const float* __restrict__ ell,
                           _Float16* __restrict__ T) {
    __shared__ _Float16 lds_a[16 * S1_STRIDE];

    const int row0 = blockIdx.x * 16;
    const int tid  = threadIdx.x;
    const int wave = tid >> 5;
    const int lane = tid & 31;
    const int rr   = lane & 15;
    const int half = lane >> 4;

    // stager mapping: thread -> (row sr, cols sc..sc+1), 256 thr = 16x32 elems
    const int sr = tid >> 4;
    const int sc = (tid & 15) * 2;

    v8f acc = {};

    for (int k0 = 0; k0 < D_IN; k0 += 32) {
        // ---- stage 16x32 (x*g) panel: b64 x-load, b64 g-load, b32 ds store --
        {
            const v2f xv = *(const v2f*)&x[(size_t)(row0 + sr) * D_IN + k0 + sc];
            const v2f gv = *(const v2f*)&g[k0 + sc];
            v2h p;
            p[0] = (_Float16)(xv[0] * gv[0]);
            p[1] = (_Float16)(xv[1] * gv[1]);
            *(v2h*)&lds_a[sr * S1_STRIDE + sc] = p;
            __builtin_prefetch(&x[(size_t)(row0 + sr) * D_IN + k0 + 32 + sc], 0, 0);
        }
        __syncthreads();

        // ---- A from LDS (2x ds_load_b128), B from swizzled global (b128 x2) -
        const v16h a = load_a_frag(lds_a, rr, S1_STRIDE, 0, half);
        const v16h b = *(const v16h*)&Vs_sw[(size_t)(((k0 >> 5) * 8 + wave) << 9)
                                            + lane * 16];

        acc = __builtin_amdgcn_wmma_f32_16x16x32_f16(
                  false, a, false, b, (short)0, acc, false, false);
        __syncthreads();
    }

    // ---- store T with ell folded in (C layout: VGPR i -> M = i + 8*half) ----
    const int   n  = wave * 16 + rr;
    const float el = ell[n];
    #pragma unroll
    for (int i = 0; i < 8; ++i) {
        int m = i + half * 8;
        T[(size_t)(row0 + m) * RANK + n] = (_Float16)(acc[i] * el);
    }
}

// ---------------------------------------------------------------------------
// Kernel 2: out = T @ sign(U)^T, fused *h + bias.
// Block = 8 waves = 16 rows x 128 cols; grid (512, 32).
// T panel (16x128 f16) staged into LDS by the Tensor Data Mover: one D# with
// pad_enable (pad 4 DWORDs after every 64 DWORDs) produces the padded
// 136-half LDS row stride directly.
// ---------------------------------------------------------------------------
#define S2_STRIDE 136  // halfs; 272B row stride (16B aligned, conflict-spread)

__global__ void lbl_stage2(const _Float16* __restrict__ T,
                           const _Float16* __restrict__ UT_sw,
                           const float* __restrict__ h,
                           const float* __restrict__ bias,
                           float* __restrict__ out) {
    __shared__ _Float16 lds_t[16 * S2_STRIDE];

    const int row0 = blockIdx.x * 16;
    const int tid  = threadIdx.x;
    const int wave = tid >> 5;
    const int lane = tid & 31;
    const int rr   = lane & 15;
    const int half = lane >> 4;
    const int cs   = blockIdx.y * 8 + wave;     // 16-col slice id, 0..255
    const int c0   = cs * 16;

#if __has_builtin(__builtin_amdgcn_tensor_load_to_lds) && \
    __has_builtin(__builtin_amdgcn_s_wait_tensorcnt)
    if (tid < 32) {                              // wave 0 issues the TDM copy
        const unsigned long long gaddr =
            (unsigned long long)(uintptr_t)(T + (size_t)row0 * RANK);
        const unsigned int ldsa = (unsigned int)(uintptr_t)(&lds_t[0]);
        v4u g0;
        g0[0] = 1u;                                       // count=1, user D#
        g0[1] = ldsa;                                     // lds_addr
        g0[2] = (unsigned int)gaddr;                      // global_addr[31:0]
        g0[3] = (unsigned int)((gaddr >> 32) & 0x1FFFFFFu)
                | (2u << 30);                             // addr[56:32] | type=2
        v8i_t g1;
        g1[0] = (int)((1u << 16)      // data_size = 2 bytes
                    | (1u << 20)      // pad_enable
                    | (5u << 22)      // pad_interval: 64 DWORDs (one 128h row)
                    | (3u << 25));    // pad_amount: 4 DWORDs (8 halfs)
        g1[1] = (int)(128u << 16);    // tensor_dim0 = 128 (bits 63:48)
        g1[2] = (int)(16u << 16);     // tensor_dim0 hi | tensor_dim1 = 16
        g1[3] = (int)(128u << 16);    // tensor_dim1 hi | tile_dim0 = 128
        g1[4] = 16;                   // tile_dim1 = 16, tile_dim2 = 0
        g1[5] = 128;                  // tensor_dim0_stride = 128 elements
        g1[6] = 0;
        g1[7] = 0;
        const v4i_t z4 = {0, 0, 0, 0};
        const v8i_t z8 = {0, 0, 0, 0, 0, 0, 0, 0};
        // 6-arg variant (clang-23 / therock-10.0 headers): groups 2, 3, spare
        __builtin_amdgcn_tensor_load_to_lds(g0, g1, z4, z4, z8, 0);
        __builtin_amdgcn_s_wait_tensorcnt(0);
    }
    __syncthreads();
#else
    // fallback: cooperative b128 copy into the padded layout
    {
        int r = tid >> 4;                // 0..15
        int c = (tid & 15) * 8;          // 8 halfs per thread
        *(v8h*)&lds_t[r * S2_STRIDE + c] =
            *(const v8h*)&T[(size_t)(row0 + r) * RANK + c];
    }
    __syncthreads();
#endif

    v8f acc = {};
    #pragma unroll
    for (int kk = 0; kk < RANK; kk += 32) {
        const v16h a = load_a_frag(lds_t, rr, S2_STRIDE, kk, half);
        const v16h b = *(const v16h*)&UT_sw[(size_t)((((kk >> 5) * 256) + cs) << 9)
                                            + lane * 16];
        acc = __builtin_amdgcn_wmma_f32_16x16x32_f16(
                  false, a, false, b, (short)0, acc, false, false);
    }

    // ---- fused epilogue: *h + bias, f32 stores ----
    const int   col = c0 + rr;
    const float hc  = h[col];
    const float bc  = bias[col];
    #pragma unroll
    for (int i = 0; i < 8; ++i) {
        int m = i + half * 8;
        out[(size_t)(row0 + m) * D_OUT + col] = acc[i] * hc + bc;
    }
}

// ---------------------------------------------------------------------------
extern "C" void kernel_launch(void* const* d_in, const int* in_sizes, int n_in,
                              void* d_out, int out_size, void* d_ws, size_t ws_size,
                              hipStream_t stream) {
    const float* x    = (const float*)d_in[0];
    const float* U_fp = (const float*)d_in[1];
    const float* V_fp = (const float*)d_in[2];
    const float* h    = (const float*)d_in[3];
    const float* g    = (const float*)d_in[4];
    const float* ell  = (const float*)d_in[5];
    const float* bias = (const float*)d_in[6];
    float* out = (float*)d_out;

    char* ws = (char*)d_ws;
    _Float16* Vs_sw = (_Float16*)(ws);                               // 1 MB
    _Float16* UT_sw = (_Float16*)(ws + (size_t)D_IN * RANK * 2);     // 1 MB
    _Float16* T     = (_Float16*)(ws + (size_t)2 * D_IN * RANK * 2); // 2 MB

    lbl_sign_convert<<<(D_OUT * RANK) / 256, 256, 0, stream>>>(U_fp, V_fp,
                                                               Vs_sw, UT_sw);
    lbl_stage1<<<ROWS / 16, 256, 0, stream>>>(x, g, Vs_sw, ell, T);
    lbl_stage2<<<dim3(ROWS / 16, D_OUT / 128), 256, 0, stream>>>(T, UT_sw, h,
                                                                 bias, out);
}
